// MultiHeadAttention_47734266528489
// MI455X (gfx1250) — matrix operand order repaired and checked by reading, not compiled, not run on hardware
//
#include <hip/hip_runtime.h>
#include <hip/hip_bf16.h>

// MHA forward for B=4, S=2048, D=512, H=8, DH=64 on gfx1250 (wave32, WMMA).
// Pipeline (all on `stream`, in order):
//   1) gemm_nt<f32in,bf16out> x3 : q = (Q Wq^T)/8, k = K Wk^T, v = V Wv^T   (bf16 in ws)
//   2) flash_attn              : ctx = softmax(q k^T) v, written as concat [B,S,D] bf16
//   3) gemm_nt<bf16in,f32out>  : out = ctx Wout^T                            (f32 d_out)
// Wq[h][e][d] flattened is exactly W[col=h*64+e][d], so every projection is the
// same NT GEMM:  Out[row,col] = sum_d In[row,d] * W[col,d].
// The additive mask where(mask==0,0,-1e30) is a per-query-row constant broadcast
// over keys; softmax is shift-invariant per row, so it cannot change the output.

#define BB 4
#define SS 2048
#define DD 512
#define HH 8
#define DHH 64

typedef __attribute__((ext_vector_type(16))) __bf16 v16bf;
typedef __attribute__((ext_vector_type(2)))  __bf16 v2bf;
typedef __attribute__((ext_vector_type(8)))  float  v8f;
typedef __attribute__((ext_vector_type(4)))  int    v4i;

union Frag16 {
  v16bf v;
  unsigned short u[16];
  unsigned int   w[8];
};

// ---- feature probes (target-aware __has_builtin per bridge doc) -------------
#if defined(__gfx1250__) && __has_builtin(__builtin_amdgcn_global_load_async_to_lds_b128) && \
    __has_builtin(__builtin_amdgcn_s_wait_asynccnt)
#define USE_ASYNC_LDS 1
#else
#define USE_ASYNC_LDS 0
#endif
#if defined(__gfx1250__) && __has_builtin(__builtin_amdgcn_cvt_pk_bf16_f32)
#define USE_CVT_PK 1
#else
#define USE_CVT_PK 0
#endif

#if USE_ASYNC_LDS
// Builtin signature (from hipcc diagnostic): param0 is a pointer to int4 in the
// global address space, param1 a pointer to int4 in LDS, then imm offset, imm cpol.
typedef __attribute__((address_space(1))) v4i gv4i;
typedef __attribute__((address_space(3))) v4i lv4i;
__device__ __forceinline__ gv4i* as_global(const void* p) {
  return (gv4i*)(unsigned long long)p;   // flat VA == global address
}
__device__ __forceinline__ lv4i* as_lds(void* p) {
  // low 32 bits of a flat LDS address are the LDS byte offset (ISA 10.2)
  return (lv4i*)(unsigned int)(unsigned long long)p;
}
#endif

__device__ __forceinline__ unsigned short f2bf(float f) {
  unsigned int x = __float_as_uint(f);
  x += 0x7FFFu + ((x >> 16) & 1u);  // round-to-nearest-even
  return (unsigned short)(x >> 16);
}
__device__ __forceinline__ unsigned int pack2bf(float lo, float hi) {
#if USE_CVT_PK
  union { v2bf v; unsigned int w; } t;
  t.v = __builtin_amdgcn_cvt_pk_bf16_f32(lo, hi);  // v_cvt_pk_bf16_f32
  return t.w;
#else
  return (unsigned int)f2bf(lo) | ((unsigned int)f2bf(hi) << 16);
#endif
}
__device__ __forceinline__ v8f wmma_bf16(v16bf a, v16bf b, v8f c) {
  // D(16x16 f32) = A(16x32 bf16) * B(32x16 bf16) + C
  return __builtin_amdgcn_wmma_f32_16x16x32_bf16(false, a, false, b, (short)0, c,
                                                 false, false);
}

// ---------------------------------------------------------------------------
// NT GEMM: Out[row,col] = scale * sum_d In[row,d] * W[col,d]
// M = B*S = 8192, N = K = 512. Block = 128 threads (4 waves); workgroup tile
// 64x64; each wave does a 16x64 strip (4 WMMA accumulators, K-loop step 32).
// ---------------------------------------------------------------------------
template <bool IN_F32, bool OUT_F32>
__global__ __launch_bounds__(128) void gemm_nt_kernel(const void* __restrict__ inp,
                                                      const float* __restrict__ Wp,
                                                      void* __restrict__ outp,
                                                      float scale) {
  const int Kd = DD, N = DD;
  const int wave = threadIdx.x >> 5, lane = threadIdx.x & 31;
  const int half = lane >> 4, nm = lane & 15;
  const int rowTile = blockIdx.y * 64, colTile = blockIdx.x * 64;
  const int arow = rowTile + wave * 16 + nm;  // A-fragment row (M = lane&15)

  const float*          inF = (const float*)inp;
  const unsigned short* inB = (const unsigned short*)inp;

  v8f acc[4] = {};

#pragma unroll 4
  for (int kk = 0; kk < Kd; kk += 32) {
    // A fragment: elem pair p -> K = kk + 16*half + 2*p
    Frag16 a;
#pragma unroll
    for (int p = 0; p < 8; ++p) {
      int d = kk + 16 * half + 2 * p;
      if (IN_F32) {
        float2 f = *(const float2*)(inF + (size_t)arow * Kd + d);
        a.w[p] = pack2bf(f.x, f.y);
      } else {
        a.w[p] = *(const unsigned int*)(inB + (size_t)arow * Kd + d);
      }
    }
#pragma unroll
    for (int nb = 0; nb < 4; ++nb) {
      // B fragment: elem pair p -> K = kk + 16*half + 2*p, column = colTile+nb*16+nm
      const int col = colTile + nb * 16 + nm;
      Frag16 bfr;
#pragma unroll
      for (int p = 0; p < 8; ++p) {
        int d = kk + 16 * half + 2 * p;
        float2 f = *(const float2*)(Wp + (size_t)col * Kd + d);
        bfr.w[p] = pack2bf(f.x, f.y);
      }
      acc[nb] = wmma_bf16(a.v, bfr.v, acc[nb]);
    }
  }

  // C/D layout: lane(half,nm) VGPR r -> [row r+8*half, col nm]
#pragma unroll
  for (int nb = 0; nb < 4; ++nb) {
#pragma unroll
    for (int r = 0; r < 8; ++r) {
      int orow = rowTile + wave * 16 + r + 8 * half;
      int ocol = colTile + nb * 16 + nm;
      float vv = acc[nb][r] * scale;
      if (OUT_F32)
        ((float*)outp)[(size_t)orow * N + ocol] = vv;
      else
        ((unsigned short*)outp)[(size_t)orow * N + ocol] = f2bf(vv);
    }
  }
}

// ---------------------------------------------------------------------------
// Flash attention. Grid = B*H*(S/64) blocks of 128 threads (4 waves).
// Each wave owns 16 query rows; the workgroup streams 32-key K/V tiles through
// LDS. q already carries the 1/sqrt(DH) scale.
//   Kt : row-major [key][d]   -> QK^T B-fragments are contiguous (ds_load_b128)
//   Vt : transposed [d][key]  -> PV  B-fragments are contiguous (ds_load_b128)
// Row sums are computed with one WMMA against a ones-matrix (row sum lands in
// every column of the C tile), replacing 32 ds_bpermute per tile.
// ---------------------------------------------------------------------------
__global__ __launch_bounds__(128) void flash_attn_kernel(
    const unsigned short* __restrict__ qb, const unsigned short* __restrict__ kp,
    const unsigned short* __restrict__ vp, unsigned short* __restrict__ cb) {
  __shared__ __align__(16) unsigned short Kt[32 * 64];      // [key][d] bf16
  __shared__ __align__(16) unsigned short Vt[64 * 32];      // [d][key] bf16
  __shared__ __align__(16) unsigned short Pt[4][16 * 32];   // wave-private P tiles

  const int bid = blockIdx.x;
  const int qt = bid & 31, h = (bid >> 5) & 7, b = bid >> 8;
  const int tid = threadIdx.x, wave = tid >> 5, lane = tid & 31;
  const int half = lane >> 4, nm = lane & 15;
  const int qrow0 = qt * 64 + wave * 16;

  // Q A-fragments for the two 32-wide K-chunks of DH=64 (loop invariant)
  Frag16 aq[2];
  {
    const size_t base = ((size_t)(b * SS + qrow0 + nm)) * DD + h * DHH;
#pragma unroll
    for (int c = 0; c < 2; ++c)
#pragma unroll
      for (int p = 0; p < 8; ++p) {
        int d = 32 * c + 16 * half + 2 * p;
        aq[c].w[p] = *(const unsigned int*)(qb + base + d);
      }
  }

  Frag16 onesf;  // B-matrix of ones for WMMA row sums
#pragma unroll
  for (int i = 0; i < 8; ++i) onesf.w[i] = 0x3F803F80u;

  v8f o[4] = {};
  float mrow[8], lrow[8];
#pragma unroll
  for (int r = 0; r < 8; ++r) { mrow[r] = -__builtin_inff(); lrow[r] = 0.0f; }

  unsigned short* pw = Pt[wave];

  for (int k0 = 0; k0 < SS; k0 += 32) {
    __syncthreads();  // previous iteration finished with Kt/Vt
    // Cooperative K/V tile load (2 x 4KB). K: straight copy. V: transpose.
#pragma unroll
    for (int t = 0; t < 2; ++t) {
      int idx = tid + t * 128;
      int j = idx >> 3, d0 = (idx & 7) * 8;
      size_t g = ((size_t)(b * SS + k0 + j)) * DD + h * DHH + d0;
#if USE_ASYNC_LDS
      __builtin_amdgcn_global_load_async_to_lds_b128(as_global(kp + g),
                                                     as_lds(&Kt[j * 64 + d0]), 0, 0);
#else
      *(uint4*)&Kt[j * 64 + d0] = *(const uint4*)(kp + g);
#endif
      union { uint4 q; unsigned short s[8]; } uv;
      uv.q = *(const uint4*)(vp + g);
#pragma unroll
      for (int e = 0; e < 8; ++e) Vt[(d0 + e) * 32 + j] = uv.s[e];
    }
    if (k0 + 32 < SS) {  // prefetch next tile (global_prefetch_b8)
      size_t g = ((size_t)(b * SS + k0 + 32 + (tid >> 2))) * DD + h * DHH;
      __builtin_prefetch(kp + g, 0, 1);
      __builtin_prefetch(vp + g, 0, 1);
    }
#if USE_ASYNC_LDS
    __builtin_amdgcn_s_wait_asynccnt(0);
#endif
    __syncthreads();

    // S = q k^T : two 16x16 f32 tiles covering 32 keys, K-dim = DH = 64
    v8f s0 = {}, s1 = {};
    {
      Frag16 bk;
#pragma unroll
      for (int p = 0; p < 8; ++p)
        bk.w[p] = *(const unsigned int*)&Kt[nm * 64 + 16 * half + 2 * p];
      s0 = wmma_bf16(aq[0].v, bk.v, s0);
#pragma unroll
      for (int p = 0; p < 8; ++p)
        bk.w[p] = *(const unsigned int*)&Kt[nm * 64 + 32 + 16 * half + 2 * p];
      s0 = wmma_bf16(aq[1].v, bk.v, s0);
#pragma unroll
      for (int p = 0; p < 8; ++p)
        bk.w[p] = *(const unsigned int*)&Kt[(16 + nm) * 64 + 16 * half + 2 * p];
      s1 = wmma_bf16(aq[0].v, bk.v, s1);
#pragma unroll
      for (int p = 0; p < 8; ++p)
        bk.w[p] = *(const unsigned int*)&Kt[(16 + nm) * 64 + 32 + 16 * half + 2 * p];
      s1 = wmma_bf16(aq[1].v, bk.v, s1);
    }

    // Online softmax. Row stats live per lane for rows r+8*half; the max
    // reduction xor-shuffles stay inside each 16-lane half of the wave.
    float alpha[8];
#pragma unroll
    for (int r = 0; r < 8; ++r) {
      float mx = fmaxf(s0[r], s1[r]);
#pragma unroll
      for (int ofs = 1; ofs < 16; ofs <<= 1) mx = fmaxf(mx, __shfl_xor(mx, ofs, 32));
      float mn = fmaxf(mrow[r], mx);
      alpha[r] = __expf(mrow[r] - mn);
      mrow[r] = mn;
      s0[r] = __expf(s0[r] - mn);
      s1[r] = __expf(s1[r] - mn);
    }

    // C-layout -> A-layout transpose of P through wave-private LDS
    // (LDS ops from the same wave execute in order; no barrier needed).
#pragma unroll
    for (int r = 0; r < 8; ++r) {
      int row = r + 8 * half;
      pw[row * 32 + nm]      = f2bf(s0[r]);
      pw[row * 32 + 16 + nm] = f2bf(s1[r]);
    }
    Frag16 ap;
#pragma unroll
    for (int p = 0; p < 8; ++p) {
      int k = 16 * half + 2 * p;
      ap.w[p] = *(const unsigned int*)&pw[nm * 32 + k];
    }

    // Row sums via one WMMA: (P * ones) puts sum_k P[row,k] in every column.
    v8f zz = {};
    v8f ls = wmma_bf16(ap.v, onesf.v, zz);
#pragma unroll
    for (int r = 0; r < 8; ++r) lrow[r] = lrow[r] * alpha[r] + ls[r];

    // O = O*alpha + P * V  (K-dim = 32 keys; 4 output column blocks of DH)
#pragma unroll
    for (int nb = 0; nb < 4; ++nb) {
#pragma unroll
      for (int r = 0; r < 8; ++r) o[nb][r] *= alpha[r];
      Frag16 bv;  // Vt is [d][key]: pair p -> keys 16*half+2p(+1), col nb*16+nm
#pragma unroll
      for (int p = 0; p < 8; ++p)
        bv.w[p] = *(const unsigned int*)&Vt[(nb * 16 + nm) * 32 + 16 * half + 2 * p];
      o[nb] = wmma_bf16(ap.v, bv.v, o[nb]);
    }
  }

  // ctx written straight into concat layout [B,S,D] (col offset h*DH), bf16
#pragma unroll
  for (int nb = 0; nb < 4; ++nb) {
#pragma unroll
    for (int r = 0; r < 8; ++r) {
      int row = qrow0 + r + 8 * half;
      float vv = o[nb][r] / lrow[r];
      cb[((size_t)(b * SS + row)) * DD + h * DHH + nb * 16 + nm] = f2bf(vv);
    }
  }
}

// ---------------------------------------------------------------------------
extern "C" void kernel_launch(void* const* d_in, const int* in_sizes, int n_in,
                              void* d_out, int out_size, void* d_ws, size_t ws_size,
                              hipStream_t stream) {
  (void)in_sizes; (void)n_in; (void)out_size; (void)ws_size;
  const float* Q  = (const float*)d_in[0];
  const float* K  = (const float*)d_in[1];
  const float* V  = (const float*)d_in[2];
  // d_in[3] = mask: per-row constant added pre-softmax -> no effect (see top).
  const float* Wq = (const float*)d_in[4];
  const float* Wk = (const float*)d_in[5];
  const float* Wv = (const float*)d_in[6];
  const float* Wo = (const float*)d_in[7];

  const size_t NELT = (size_t)BB * SS * DD;  // 4M elems
  unsigned short* qb = (unsigned short*)d_ws;        // 8 MB each
  unsigned short* kb = qb + NELT;
  unsigned short* vb = kb + NELT;
  unsigned short* cb = vb + NELT;                    // total 32 MB of ws

  dim3 blk(128);
  dim3 ggrid(DD / 64, (BB * SS) / 64);  // 8 x 128 workgroups

  gemm_nt_kernel<true,  false><<<ggrid, blk, 0, stream>>>(Q, Wq, qb, 0.125f); // 1/sqrt(64)
  gemm_nt_kernel<true,  false><<<ggrid, blk, 0, stream>>>(K, Wk, kb, 1.0f);
  gemm_nt_kernel<true,  false><<<ggrid, blk, 0, stream>>>(V, Wv, vb, 1.0f);
  flash_attn_kernel<<<dim3(BB * HH * (SS / 64)), blk, 0, stream>>>(qb, kb, vb, cb);
  gemm_nt_kernel<false, true ><<<ggrid, blk, 0, stream>>>(cb, Wo, (float*)d_out, 1.0f);
}
